// RNNEncoder_27419071218035
// MI455X (gfx1250) — matrix operand order repaired and checked by reading, not compiled, not run on hardware
//
#include <hip/hip_runtime.h>
#include <math.h>

#define Hd    2048
#define OUTD  512
#define TSTEPS 1024
#define NBLK  128
#define NTHR  256
#define NWAVE 8

typedef __attribute__((ext_vector_type(16))) __bf16 v16bf;
typedef __attribute__((ext_vector_type(8)))  float  v8f;

__device__ __forceinline__ v16bf mkfrag(uint4 lo, uint4 hi) {
    union { uint4 q[2]; v16bf v; } u;
    u.q[0] = lo; u.q[1] = hi;
    return u.v;
}

__device__ __forceinline__ unsigned short f2bf(float f) {
    unsigned int u = __float_as_uint(f);
    u += 0x7fffu + ((u >> 16) & 1u);          // round-to-nearest-even
    return (unsigned short)(u >> 16);
}

// ---------------------------------------------------------------- converters
__global__ void cvt_f32_bf16(const float* __restrict__ src,
                             unsigned short* __restrict__ dst, int n) {
    int i = blockIdx.x * blockDim.x + threadIdx.x;
    int stride = gridDim.x * blockDim.x;
    for (; i < n; i += stride) dst[i] = f2bf(src[i]);
}

__global__ void init_misc(const float* __restrict__ hidden,
                          unsigned short* __restrict__ hid_bf,
                          unsigned short* __restrict__ zero_bf,
                          unsigned int* barCnt, unsigned int* barRel) {
    int i = blockIdx.x * blockDim.x + threadIdx.x;
    if (i < Hd) { hid_bf[i] = f2bf(hidden[i]); zero_bf[i] = 0; }
    if (i == 0) { *barCnt = 0u; *barRel = 0u; }
}

// ------------------------------------------------------------- persistent RNN
__global__ __launch_bounds__(NTHR, 1) void rnn_persistent(
    const unsigned short* __restrict__ Wih,   // [3H , H] bf16 row-major
    const unsigned short* __restrict__ Whh,   // [3H , H]
    const unsigned short* __restrict__ Wl4,   // [4, H, H]  (W1..W4)
    const unsigned short* __restrict__ Wo,    // [OUT, H]
    const float* __restrict__ b_ih, const float* __restrict__ b_hh,
    const float* __restrict__ b1,   const float* __restrict__ b2,
    const float* __restrict__ b3,   const float* __restrict__ b4,
    const float* __restrict__ bo,
    const float* __restrict__ hidden0,            // [H] f32
    const unsigned short* __restrict__ hid_bf,    // [H] bf16
    const unsigned short* __restrict__ zero_bf,   // [H] bf16 zeros
    float* __restrict__ hnewF,                    // [2][H]
    unsigned short* __restrict__ hnewBf,          // [2][H]
    unsigned short* __restrict__ hsBf,            // [2][H]
    float* __restrict__ out,                      // [T][OUT]
    unsigned int* barCnt, unsigned int* barRel)
{
    __shared__ unsigned int ldsX[Hd / 2];     // x vector, bf16 pairs
    __shared__ unsigned int ldsH[Hd / 2];     // h / hs vector, bf16 pairs
    __shared__ float part[6 * NWAVE * 16];    // per-wave partial dot products
    __shared__ float red[6 * 16];             // reduced gate values

    const int tid    = threadIdx.x;
    const int wave   = tid >> 5;
    const int lane   = tid & 31;
    const int laneHi = lane >> 4;
    const int laneN  = lane & 15;
    const int tile   = blockIdx.x;            // one 16-row output tile per WG
    const int k0     = wave * 256;            // K-slice per wave
    unsigned int phase = 0;

    // device-wide barrier (counter + monotone release word)
    auto gridSync = [&]() {
        __syncthreads();
        if (tid == 0) {
            ++phase;
            __threadfence();
            unsigned int prev = __hip_atomic_fetch_add(
                barCnt, 1u, __ATOMIC_ACQ_REL, __HIP_MEMORY_SCOPE_AGENT);
            if (prev == (unsigned)gridDim.x - 1u) {
                __hip_atomic_store(barCnt, 0u, __ATOMIC_RELAXED,
                                   __HIP_MEMORY_SCOPE_AGENT);
                __hip_atomic_store(barRel, phase, __ATOMIC_RELEASE,
                                   __HIP_MEMORY_SCOPE_AGENT);
            } else {
                while (__hip_atomic_load(barRel, __ATOMIC_ACQUIRE,
                                         __HIP_MEMORY_SCOPE_AGENT) < phase)
                    __builtin_amdgcn_s_sleep(2);
            }
            __threadfence();
        }
        __syncthreads();
    };

    auto copyVec = [&](unsigned int* dst, const unsigned short* src) {
        const unsigned int* s = reinterpret_cast<const unsigned int*>(src);
        for (int i = tid; i < Hd / 2; i += NTHR) dst[i] = s[i];
    };

    // Two row-tiles at once with interleaved accumulator chains.
    // B frag: lane reads bf16 {0..7,16..23} (+8 for lane>>4) of weight row
    //         (lane&15) -> two global_load_b128 per tile per chunk.
    // A frag: broadcast vector chunk; low lanes pairs {0..3,8..11},
    //         high lanes {4..7,12..15} -> two ds_load_b128 per stream.
    auto matvecPair = [&](const unsigned short* __restrict__ Wa,
                          const unsigned short* __restrict__ Wb,
                          const unsigned int* lva, const unsigned int* lvb,
                          float& ya, float& yb) {
        v8f accA = {0.f, 0.f, 0.f, 0.f, 0.f, 0.f, 0.f, 0.f};
        v8f accB = {0.f, 0.f, 0.f, 0.f, 0.f, 0.f, 0.f, 0.f};
        const unsigned short* ra = Wa + (size_t)laneN * Hd + k0 + laneHi * 8;
        const unsigned short* rb = Wb + (size_t)laneN * Hd + k0 + laneHi * 8;
#pragma unroll
        for (int c = 0; c < 8; ++c) {
            const int pb = (k0 + c * 32) >> 1;  // pair base in LDS dwords
            uint4 aa0 = *reinterpret_cast<const uint4*>(&lva[pb + laneHi * 4]);
            uint4 aa1 = *reinterpret_cast<const uint4*>(&lva[pb + 8 + laneHi * 4]);
            uint4 ab0 = *reinterpret_cast<const uint4*>(&lvb[pb + laneHi * 4]);
            uint4 ab1 = *reinterpret_cast<const uint4*>(&lvb[pb + 8 + laneHi * 4]);
            uint4 ba0 = *reinterpret_cast<const uint4*>(ra);
            uint4 ba1 = *reinterpret_cast<const uint4*>(ra + 16);
            uint4 bb0 = *reinterpret_cast<const uint4*>(rb);
            uint4 bb1 = *reinterpret_cast<const uint4*>(rb + 16);
            ra += 32; rb += 32;
            accA = __builtin_amdgcn_wmma_f32_16x16x32_bf16(
                false, mkfrag(aa0, aa1), false, mkfrag(ba0, ba1),
                (short)0, accA, false, false);
            accB = __builtin_amdgcn_wmma_f32_16x16x32_bf16(
                false, mkfrag(ab0, ab1), false, mkfrag(bb0, bb1),
                (short)0, accB, false, false);
        }
        ya = accA[0];                     // D row m=0 -> VGPR0 of lanes 0..15
        yb = accB[0];
    };

    // Single row-tile, two accumulators over even/odd k-chunks.
    auto matvecDual = [&](const unsigned short* __restrict__ W,
                          const unsigned int* lv) -> float {
        v8f acc0 = {0.f, 0.f, 0.f, 0.f, 0.f, 0.f, 0.f, 0.f};
        v8f acc1 = {0.f, 0.f, 0.f, 0.f, 0.f, 0.f, 0.f, 0.f};
        const unsigned short* r = W + (size_t)laneN * Hd + k0 + laneHi * 8;
#pragma unroll
        for (int c = 0; c < 8; c += 2) {
            const int pb0 = (k0 + c * 32) >> 1;
            const int pb1 = (k0 + (c + 1) * 32) >> 1;
            uint4 a00 = *reinterpret_cast<const uint4*>(&lv[pb0 + laneHi * 4]);
            uint4 a01 = *reinterpret_cast<const uint4*>(&lv[pb0 + 8 + laneHi * 4]);
            uint4 a10 = *reinterpret_cast<const uint4*>(&lv[pb1 + laneHi * 4]);
            uint4 a11 = *reinterpret_cast<const uint4*>(&lv[pb1 + 8 + laneHi * 4]);
            uint4 b00 = *reinterpret_cast<const uint4*>(r);
            uint4 b01 = *reinterpret_cast<const uint4*>(r + 16);
            uint4 b10 = *reinterpret_cast<const uint4*>(r + 32);
            uint4 b11 = *reinterpret_cast<const uint4*>(r + 48);
            r += 64;
            acc0 = __builtin_amdgcn_wmma_f32_16x16x32_bf16(
                false, mkfrag(a00, a01), false, mkfrag(b00, b01),
                (short)0, acc0, false, false);
            acc1 = __builtin_amdgcn_wmma_f32_16x16x32_bf16(
                false, mkfrag(a10, a11), false, mkfrag(b10, b11),
                (short)0, acc1, false, false);
        }
        return acc0[0] + acc1[0];
    };

    for (int t = 0; t < TSTEPS; ++t) {
        const int pp = t & 1;
        const unsigned short* xbf = t ? hnewBf + ((t - 1) & 1) * Hd : zero_bf;
        const unsigned short* hbf = t ? hnewBf + ((t - 1) & 1) * Hd : hid_bf;
        const float*        hprev = t ? hnewF + ((t - 1) & 1) * Hd : hidden0;

        // ---------------- Stage A: GRU gates + state update ----------------
        copyVec(ldsX, xbf);
        copyVec(ldsH, hbf);
        __syncthreads();
        {
            const size_t tb = (size_t)tile * 16 * Hd;
            const size_t ms = (size_t)Hd * Hd;    // one gate matrix
            // pairs: p0 = (iR,iZ) [x,x], p1 = (hR,hZ) [h,h], p2 = (iN,hN) [x,h]
            // unroll 1: keep exactly one pair's WMMA state live -> no spills
#pragma unroll 1
            for (int p = 0; p < 3; ++p) {
                const unsigned short* Wa =
                    (p == 0) ? Wih + tb : (p == 1) ? Whh + tb
                                        : Wih + 2 * ms + tb;
                const unsigned short* Wb =
                    (p == 0) ? Wih + ms + tb : (p == 1) ? Whh + ms + tb
                                             : Whh + 2 * ms + tb;
                const unsigned int* lva = (p == 1) ? ldsH : ldsX;
                const unsigned int* lvb = (p == 0) ? ldsX : ldsH;
                const int ra = (p == 0) ? 0 : (p == 1) ? 3 : 2;
                const int rb = (p == 0) ? 1 : (p == 1) ? 4 : 5;
                float ya, yb;
                matvecPair(Wa, Wb, lva, lvb, ya, yb);
                if (lane < 16) {
                    part[(ra * NWAVE + wave) * 16 + lane] = ya;
                    part[(rb * NWAVE + wave) * 16 + lane] = yb;
                }
            }
        }
        __syncthreads();
        if (tid < 96) {                        // reduce 8 k-slices
            const int rt = tid >> 4, n = tid & 15;
            float s = 0.f;
#pragma unroll
            for (int w = 0; w < NWAVE; ++w) s += part[(rt * NWAVE + w) * 16 + n];
            red[rt * 16 + n] = s;
        }
        __syncthreads();
        if (tid < 16) {                        // gate math, f32
            const int j = tile * 16 + tid;
            float giR = red[0 * 16 + tid] + b_ih[j];
            float giZ = red[1 * 16 + tid] + b_ih[Hd + j];
            float giN = red[2 * 16 + tid] + b_ih[2 * Hd + j];
            float ghR = red[3 * 16 + tid] + b_hh[j];
            float ghZ = red[4 * 16 + tid] + b_hh[Hd + j];
            float ghN = red[5 * 16 + tid] + b_hh[2 * Hd + j];
            float r  = 1.f / (1.f + __expf(-(giR + ghR)));
            float z  = 1.f / (1.f + __expf(-(giZ + ghZ)));
            float nn = tanhf(giN + r * ghN);
            float hn = (1.f - z) * nn + z * hprev[j];
            hnewF[pp * Hd + j]  = hn;
            hnewBf[pp * Hd + j] = f2bf(hn);
        }
        gridSync();

        // ---------------- Stage B: 4 hidden->hidden linears ----------------
        for (int l = 0; l < 4; ++l) {
            const unsigned short* vin =
                (l == 0) ? hnewBf + pp * Hd : hsBf + ((l - 1) & 1) * Hd;
            copyVec(ldsH, vin);
            __syncthreads();
            const unsigned short* W = Wl4 + ((size_t)l * Hd + tile * 16) * Hd;
            float y = matvecDual(W, ldsH);
            if (lane < 16) part[wave * 16 + lane] = y;
            __syncthreads();
            if (tid < 16) {
                float s = 0.f;
#pragma unroll
                for (int w = 0; w < NWAVE; ++w) s += part[w * 16 + tid];
                const float* bl = (l == 0) ? b1 : (l == 1) ? b2
                                 : (l == 2) ? b3 : b4;
                hsBf[(l & 1) * Hd + tile * 16 + tid] =
                    f2bf(s + bl[tile * 16 + tid]);
            }
            gridSync();
        }

        // ---------------- Stage C: output head (no trailing barrier:
        // next write of hsBf[1] is 2 barriers away in step t+1) -------------
        if (tile < OUTD / 16) {
            copyVec(ldsH, hsBf + 1 * Hd);
            __syncthreads();
            const unsigned short* W = Wo + (size_t)(tile * 16) * Hd;
            float y = matvecDual(W, ldsH);
            if (lane < 16) part[wave * 16 + lane] = y;
            __syncthreads();
            if (tid < 16) {
                float s = 0.f;
#pragma unroll
                for (int w = 0; w < NWAVE; ++w) s += part[w * 16 + tid];
                out[(size_t)t * OUTD + tile * 16 + tid] =
                    s + bo[tile * 16 + tid];
            }
        }
    }
}

// ------------------------------------------------------------------- launch
extern "C" void kernel_launch(void* const* d_in, const int* in_sizes, int n_in,
                              void* d_out, int out_size, void* d_ws,
                              size_t ws_size, hipStream_t stream) {
    const float* hidden = (const float*)d_in[0];
    // d_in[1] = target: only its length matters (T=1024), never read
    const float* W_ih = (const float*)d_in[2];
    const float* W_hh = (const float*)d_in[3];
    const float* b_ih = (const float*)d_in[4];
    const float* b_hh = (const float*)d_in[5];
    const float* W1 = (const float*)d_in[6];   const float* b1 = (const float*)d_in[7];
    const float* W2 = (const float*)d_in[8];   const float* b2 = (const float*)d_in[9];
    const float* W3 = (const float*)d_in[10];  const float* b3 = (const float*)d_in[11];
    const float* W4 = (const float*)d_in[12];  const float* b4 = (const float*)d_in[13];
    const float* Wo = (const float*)d_in[14];  const float* bo = (const float*)d_in[15];
    float* out = (float*)d_out;

    char* ws = (char*)d_ws;
    size_t off = 0;
    auto alloc = [&](size_t bytes) -> void* {
        off = (off + 255) & ~(size_t)255;
        void* p = ws + off;
        off += bytes;
        return p;
    };

    unsigned short* Wih_bf = (unsigned short*)alloc((size_t)3 * Hd * Hd * 2);
    unsigned short* Whh_bf = (unsigned short*)alloc((size_t)3 * Hd * Hd * 2);
    unsigned short* Wl4_bf = (unsigned short*)alloc((size_t)4 * Hd * Hd * 2);
    unsigned short* Wo_bf  = (unsigned short*)alloc((size_t)OUTD * Hd * 2);
    unsigned short* hid_bf  = (unsigned short*)alloc(Hd * 2);
    unsigned short* zero_bf = (unsigned short*)alloc(Hd * 2);
    float*          hnewF   = (float*)alloc(2 * Hd * 4);
    unsigned short* hnewBf  = (unsigned short*)alloc(2 * Hd * 2);
    unsigned short* hsBf    = (unsigned short*)alloc(2 * Hd * 2);
    unsigned int*   barCnt  = (unsigned int*)alloc(4);
    unsigned int*   barRel  = (unsigned int*)alloc(4);
    if (off > ws_size) return;   // workspace too small: bail deterministically

    // one-time (per launch) fp32 -> bf16 weight conversion; resident in L2
    cvt_f32_bf16<<<1024, 256, 0, stream>>>(W_ih, Wih_bf, 3 * Hd * Hd);
    cvt_f32_bf16<<<1024, 256, 0, stream>>>(W_hh, Whh_bf, 3 * Hd * Hd);
    cvt_f32_bf16<<<1024, 256, 0, stream>>>(W1, Wl4_bf + (size_t)0 * Hd * Hd, Hd * Hd);
    cvt_f32_bf16<<<1024, 256, 0, stream>>>(W2, Wl4_bf + (size_t)1 * Hd * Hd, Hd * Hd);
    cvt_f32_bf16<<<1024, 256, 0, stream>>>(W3, Wl4_bf + (size_t)2 * Hd * Hd, Hd * Hd);
    cvt_f32_bf16<<<1024, 256, 0, stream>>>(W4, Wl4_bf + (size_t)3 * Hd * Hd, Hd * Hd);
    cvt_f32_bf16<<<1024, 256, 0, stream>>>(Wo, Wo_bf, OUTD * Hd);
    init_misc<<<(Hd + 255) / 256, 256, 0, stream>>>(hidden, hid_bf, zero_bf,
                                                    barCnt, barRel);

    rnn_persistent<<<NBLK, NTHR, 0, stream>>>(
        Wih_bf, Whh_bf, Wl4_bf, Wo_bf,
        b_ih, b_hh, b1, b2, b3, b4, bo,
        hidden, hid_bf, zero_bf,
        hnewF, hnewBf, hsBf, out, barCnt, barRel);

    (void)in_sizes; (void)n_in; (void)out_size;
}
